// PointNet2FeatureDecoder_26877905338428
// MI455X (gfx1250) — matrix operand order repaired and checked by reading, not compiled, not run on hardware
//
#include <hip/hip_runtime.h>
#include <stdint.h>

// ---------------------------------------------------------------------------
// PointNet++ feature decoder for MI455X (gfx1250, wave32, WMMA).
//
// Roofline: ~32 GFLOP of GEMM over <70MB of L2-resident tensors => the GEMMs
// are latency/bandwidth bound, not math bound. Strategy:
//   * 3-NN search streamed per target point (never materialize [B,N,S] dists)
//   * activations row-major [B*N, C] f16; every 1x1 conv = one flat GEMM on
//     v_wmma_f32_16x16x32_f16 (f32 accumulate)
//   * GEMM: block stages the whole f16 weight matrix (256 x K, 128-176KB)
//     into LDS via CDNA5 async-to-LDS DMA (global_load_async_to_lds_b128,
//     ASYNCcnt); each wave owns a 16x256 output strip: A fragment loaded once
//     per k-step (double-buffered), B fragments double-buffered from LDS so
//     ds_load overlaps the WMMA burst.  __launch_bounds__(256,1) so the 128
//     accumulator VGPRs stay in registers (round-2 asm showed scratch spills
//     at the default occupancy-driven VGPR cap).
//   * conv bias skipped: training-mode BN cancels per-channel constants
//   * BN = per-channel reduction kernel + fused scale/shift(+ReLU)+f16 pack
// ---------------------------------------------------------------------------

typedef _Float16 h16;
typedef __attribute__((ext_vector_type(16))) _Float16 v16h;
typedef __attribute__((ext_vector_type(8)))  float    v8f;
typedef __attribute__((ext_vector_type(4)))  float    v4f;

constexpr int kB  = 8;
constexpr int kN0 = 8192;
constexpr int kN1 = 2048;
constexpr int kN2 = 512;
constexpr int kC  = 256;     // conv output channels (all layers)
constexpr int kCin0 = 352;   // 96 skip + 256 interpolated (fp2 first layer)
constexpr int kM1 = kB * kN1;   // 16384 rows for fp2 GEMMs
constexpr int kM0 = kB * kN0;   // 65536 rows for fp1 GEMMs
constexpr float kEpsBN = 1e-5f;

// ---------------------------------------------------------------------------
// CDNA5 async global->LDS copy (16B per lane), tracked by ASYNCcnt.
// ---------------------------------------------------------------------------
__device__ __forceinline__ void async_copy_b128(uint32_t lds_byte_off,
                                                const void* gptr) {
  asm volatile("global_load_async_to_lds_b128 %0, %1, off"
               :: "v"(lds_byte_off), "v"((uint64_t)(uintptr_t)gptr)
               : "memory");
}
__device__ __forceinline__ void wait_asynccnt0() {
  asm volatile("s_wait_asynccnt 0x0" ::: "memory");
}

// ---------------------------------------------------------------------------
// f32 -> f16 weight conversion (weights are tiny; done once per launch)
// ---------------------------------------------------------------------------
__global__ void f32_to_f16_kernel(const float* __restrict__ src,
                                  h16* __restrict__ dst, int n) {
  int i = blockIdx.x * blockDim.x + threadIdx.x;
  if (i < n) dst[i] = (h16)src[i];
}

// ---------------------------------------------------------------------------
// fp2 interpolation: for each l1 point, 3-NN among l2 points, inverse-distance
// weighted blend of l2_points [B,256,N2] (channel-major), concatenated after
// the 96 skip channels from l1_points.  Output X2: [B*N1, 352] f16 row-major.
// ---------------------------------------------------------------------------
__global__ void interp2_kernel(const float* __restrict__ l1_xyz,
                               const float* __restrict__ l2_xyz,
                               const float* __restrict__ l1_points,
                               const float* __restrict__ l2_points,
                               h16* __restrict__ X2) {
  int p = blockIdx.x * blockDim.x + threadIdx.x;
  if (p >= kM1) return;
  int b = p / kN1, n = p - b * kN1;

  const float* x1 = l1_xyz + (size_t)b * 3 * kN1 + n;
  float px = x1[0], py = x1[kN1], pz = x1[2 * kN1];
  const float* x2 = l2_xyz + (size_t)b * 3 * kN2;

  float d0 = 3.4e38f, d1 = 3.4e38f, d2 = 3.4e38f;
  int i0 = 0, i1 = 0, i2 = 0;
#pragma unroll 4
  for (int s = 0; s < kN2; ++s) {
    float dx = px - x2[s], dy = py - x2[kN2 + s], dz = pz - x2[2 * kN2 + s];
    float d = dx * dx + dy * dy + dz * dz;
    if (d < d0)      { d2 = d1; i2 = i1; d1 = d0; i1 = i0; d0 = d; i0 = s; }
    else if (d < d1) { d2 = d1; i2 = i1; d1 = d;  i1 = s; }
    else if (d < d2) { d2 = d;  i2 = s; }
  }
  float w0 = 1.f / (d0 + 1e-8f), w1 = 1.f / (d1 + 1e-8f), w2 = 1.f / (d2 + 1e-8f);
  float inv = 1.f / (w0 + w1 + w2);
  w0 *= inv; w1 *= inv; w2 *= inv;

  h16* xr = X2 + (size_t)p * kCin0;
  const float* sk = l1_points + (size_t)b * 96 * kN1 + n;   // [b,c,n], stride N1
#pragma unroll 4
  for (int c = 0; c < 96; ++c) xr[c] = (h16)sk[(size_t)c * kN1];
  const float* f = l2_points + (size_t)b * kC * kN2;        // [b,c,s], stride N2
#pragma unroll 4
  for (int c = 0; c < kC; ++c) {
    const float* fc = f + (size_t)c * kN2;
    xr[96 + c] = (h16)(w0 * fc[i0] + w1 * fc[i1] + w2 * fc[i2]);
  }
}

// ---------------------------------------------------------------------------
// fp1 interpolation: for each l0 point, 3-NN among l1 points, blend the f16
// l1p features (row-major [B*N1, 256]).  Output X0: [B*N0, 256] f16.
// ---------------------------------------------------------------------------
__global__ void interp1_kernel(const float* __restrict__ l0_xyz,
                               const float* __restrict__ l1_xyz,
                               const h16* __restrict__ feat,
                               h16* __restrict__ X0) {
  int p = blockIdx.x * blockDim.x + threadIdx.x;
  if (p >= kM0) return;
  int b = p / kN0, n = p - b * kN0;

  const float* x1 = l0_xyz + (size_t)b * 3 * kN0 + n;
  float px = x1[0], py = x1[kN0], pz = x1[2 * kN0];
  const float* x2 = l1_xyz + (size_t)b * 3 * kN1;

  float d0 = 3.4e38f, d1 = 3.4e38f, d2 = 3.4e38f;
  int i0 = 0, i1 = 0, i2 = 0;
#pragma unroll 4
  for (int s = 0; s < kN1; ++s) {
    float dx = px - x2[s], dy = py - x2[kN1 + s], dz = pz - x2[2 * kN1 + s];
    float d = dx * dx + dy * dy + dz * dz;
    if (d < d0)      { d2 = d1; i2 = i1; d1 = d0; i1 = i0; d0 = d; i0 = s; }
    else if (d < d1) { d2 = d1; i2 = i1; d1 = d;  i1 = s; }
    else if (d < d2) { d2 = d;  i2 = s; }
  }
  float w0 = 1.f / (d0 + 1e-8f), w1 = 1.f / (d1 + 1e-8f), w2 = 1.f / (d2 + 1e-8f);
  float inv = 1.f / (w0 + w1 + w2);
  w0 *= inv; w1 *= inv; w2 *= inv;

  const h16* fb = feat + (size_t)b * kN1 * kC;
  const h16* r0 = fb + (size_t)i0 * kC;
  const h16* r1 = fb + (size_t)i1 * kC;
  const h16* r2 = fb + (size_t)i2 * kC;
  h16* o = X0 + (size_t)p * kC;
#pragma unroll 8
  for (int c = 0; c < kC; ++c)
    o[c] = (h16)(w0 * (float)r0[c] + w1 * (float)r1[c] + w2 * (float)r2[c]);
}

// ---------------------------------------------------------------------------
// GEMM:  Y[M,256] (f32) = X[M,K] (f16) * W[256,K]^T (f16), f32 accumulate.
//
// Block = 256 threads = 8 waves, covers a 128-row strip of M and ALL 256 cols.
// Step 1: block DMAs the full W (256 x K f16) into LDS with
//         global_load_async_to_lds_b128 (ASYNCcnt), then s_wait_asynccnt+bar.
// Step 2: each wave owns one 16-row m-tile and all 16 n-tiles:
//         16 x v8f accumulators, A fragment loaded once per k-step from
//         global (double-buffered in regs), B fragments double-buffered
//         from LDS so ds_load overlaps WMMA.
//
// Fragment layouts per CDNA5 ISA 7.12.2 (wave32):
//   A: lane row = lane%16; elems 0..7 = K[k0+8*hi..], 8..15 = K[k0+16+8*hi..]
//   B: lane row (output ch) = lane%16; elems 0..7 = K[k0+8*hi..], 8..15 = K[k0+16+8*hi..]
//   C/D: vgpr r -> row 8*hi+r, col lane%16
// ---------------------------------------------------------------------------
__global__ void __launch_bounds__(256, 1)
gemm_wmma_f16_kernel(const h16* __restrict__ X,
                     const h16* __restrict__ W,
                     float* __restrict__ Y,
                     int M, int K) {
  extern __shared__ h16 smem[];           // [256][K] f16 weight tile

  // ---- stage W into LDS: thread t copies W row t via 16B async DMAs ----
  {
    int t = threadIdx.x;                  // 256 threads == 256 weight rows
    const h16* wrow = W + (size_t)t * K;
    h16* srow = smem + (size_t)t * K;
    for (int k = 0; k < K; k += 8)
      async_copy_b128((uint32_t)(uintptr_t)&srow[k], wrow + k);
  }
  wait_asynccnt0();
  __syncthreads();

  int lane = threadIdx.x & 31;
  int wave = threadIdx.x >> 5;
  int mt = blockIdx.x * 8 + wave;         // m-tile (16 rows) owned by wave
  int lo = lane & 15;
  int hi = lane >> 4;

  const h16* arow = X + (size_t)(mt * 16 + lo) * K;

  union Frag { v16h h; v4f f[2]; };
  v8f acc[16];
#pragma unroll
  for (int nt = 0; nt < 16; ++nt) acc[nt] = (v8f){};

  Frag a_cur, a_nxt;
  a_cur.f[0] = *(const v4f*)(arow + 8 * hi);
  a_cur.f[1] = *(const v4f*)(arow + 8 * hi + 16);
  a_nxt = a_cur;

  for (int k0 = 0; k0 < K; k0 += 32) {
    if (k0 + 32 < K) {                    // prefetch next A fragment (regs)
      const h16* pa = arow + k0 + 32 + 8 * hi;
      a_nxt.f[0] = *(const v4f*)(pa);
      a_nxt.f[1] = *(const v4f*)(pa + 16);
    }
    // B fragment double-buffer: load nt+1 while WMMA consumes nt.
    Frag b_cur, b_nxt;
    {
      const h16* pb = &smem[(size_t)lo * K + k0 + 8 * hi];
      b_cur.f[0] = *(const v4f*)(pb);
      b_cur.f[1] = *(const v4f*)(pb + 16);
    }
    b_nxt = b_cur;
#pragma unroll
    for (int nt = 0; nt < 16; ++nt) {
      if (nt < 15) {
        const h16* pb = &smem[(size_t)((nt + 1) * 16 + lo) * K + k0 + 8 * hi];
        b_nxt.f[0] = *(const v4f*)(pb);
        b_nxt.f[1] = *(const v4f*)(pb + 16);
      }
      acc[nt] = __builtin_amdgcn_wmma_f32_16x16x32_f16(
          false, a_cur.h, false, b_cur.h, (short)0, acc[nt], false, false);
      b_cur = b_nxt;
    }
    a_cur = a_nxt;
  }

  float* outb = Y + (size_t)(mt * 16 + 8 * hi) * kC + lo;
#pragma unroll
  for (int nt = 0; nt < 16; ++nt)
#pragma unroll
    for (int r = 0; r < 8; ++r)
      outb[(size_t)r * kC + nt * 16] = acc[nt][r];
}

// ---------------------------------------------------------------------------
// Training-mode BN stats over M rows per channel: scale/shift precompute.
// One block (256 threads) per channel; LDS tree reduction.
// ---------------------------------------------------------------------------
__global__ void bn_stats_kernel(const float* __restrict__ Y,
                                const float* __restrict__ gamma,
                                const float* __restrict__ beta,
                                float* __restrict__ ss, int M) {
  __shared__ float s_sum[256];
  __shared__ float s_sq[256];
  int c = blockIdx.x, t = threadIdx.x;
  float s = 0.f, q = 0.f;
  for (int m = t; m < M; m += 256) {
    float v = Y[(size_t)m * kC + c];
    s += v; q += v * v;
  }
  s_sum[t] = s; s_sq[t] = q;
  __syncthreads();
  for (int off = 128; off > 0; off >>= 1) {
    if (t < off) { s_sum[t] += s_sum[t + off]; s_sq[t] += s_sq[t + off]; }
    __syncthreads();
  }
  if (t == 0) {
    float mean = s_sum[0] / (float)M;
    float var  = s_sq[0] / (float)M - mean * mean;
    float scale = gamma[c] * rsqrtf(var + kEpsBN);
    ss[2 * c]     = scale;
    ss[2 * c + 1] = beta[c] - mean * scale;
  }
}

// BN apply + ReLU, packing f16 activations for the next GEMM's A operand.
__global__ void bn_relu_f16_kernel(const float* __restrict__ Y,
                                   const float* __restrict__ ss,
                                   h16* __restrict__ out, int M) {
  size_t i = (size_t)blockIdx.x * blockDim.x + threadIdx.x;
  if (i >= (size_t)M * kC) return;
  int c = (int)(i & (kC - 1));
  float v = Y[i] * ss[2 * c] + ss[2 * c + 1];
  out[i] = (h16)fmaxf(v, 0.f);
}

// BN apply + ReLU, f32 output (pre-final-BN activation; in-place safe).
__global__ void bn_relu_f32_kernel(const float* __restrict__ Y,
                                   const float* __restrict__ ss,
                                   float* __restrict__ out, int M) {
  size_t i = (size_t)blockIdx.x * blockDim.x + threadIdx.x;
  if (i >= (size_t)M * kC) return;
  int c = (int)(i & (kC - 1));
  float v = Y[i] * ss[2 * c] + ss[2 * c + 1];
  out[i] = fmaxf(v, 0.f);
}

// Final BN (no ReLU) -> d_out (already in [B,N0,256] row-major = ref output).
__global__ void bn_final_kernel(const float* __restrict__ Y,
                                const float* __restrict__ ss,
                                float* __restrict__ out, int M) {
  size_t i = (size_t)blockIdx.x * blockDim.x + threadIdx.x;
  if (i >= (size_t)M * kC) return;
  int c = (int)(i & (kC - 1));
  out[i] = Y[i] * ss[2 * c] + ss[2 * c + 1];
}

// Second output: l0_xyz transposed to [B, N0, 3].
__global__ void xyz_transpose_kernel(const float* __restrict__ l0_xyz,
                                     float* __restrict__ out) {
  int i = blockIdx.x * blockDim.x + threadIdx.x;
  if (i >= kM0 * 3) return;
  int d = i % 3;
  int p = i / 3;
  int b = p / kN0, n = p - b * kN0;
  out[i] = l0_xyz[((size_t)b * 3 + d) * kN0 + n];
}

// ---------------------------------------------------------------------------
extern "C" void kernel_launch(void* const* d_in, const int* in_sizes, int n_in,
                              void* d_out, int out_size, void* d_ws, size_t ws_size,
                              hipStream_t stream) {
  const float* l0_xyz    = (const float*)d_in[0];
  const float* l1_xyz    = (const float*)d_in[1];
  const float* l2_xyz    = (const float*)d_in[2];
  const float* l1_points = (const float*)d_in[3];
  const float* l2_points = (const float*)d_in[4];

  // params dict leaves: detect flatten order from sizes.
  // sorted keys  : bn(g,be), fp1[3x(w,b,g,be)], fp2[2x(w,b,g,be)] -> in_sizes[5]==256
  // insert order : fp2[2x(w,b,g,be)], fp1[3x(w,b,g,be)], bn(g,be) -> in_sizes[5]==90112
  const float *W[5], *G[6], *Be[6];
  if (n_in >= 27 && in_sizes[5] == 256) {        // sorted-key pytree order
    G[5] = (const float*)d_in[5];  Be[5] = (const float*)d_in[6];
    W[2] = (const float*)d_in[7];  G[2] = (const float*)d_in[9];  Be[2] = (const float*)d_in[10];
    W[3] = (const float*)d_in[11]; G[3] = (const float*)d_in[13]; Be[3] = (const float*)d_in[14];
    W[4] = (const float*)d_in[15]; G[4] = (const float*)d_in[17]; Be[4] = (const float*)d_in[18];
    W[0] = (const float*)d_in[19]; G[0] = (const float*)d_in[21]; Be[0] = (const float*)d_in[22];
    W[1] = (const float*)d_in[23]; G[1] = (const float*)d_in[25]; Be[1] = (const float*)d_in[26];
  } else {                                       // insertion order
    W[0] = (const float*)d_in[5];  G[0] = (const float*)d_in[7];  Be[0] = (const float*)d_in[8];
    W[1] = (const float*)d_in[9];  G[1] = (const float*)d_in[11]; Be[1] = (const float*)d_in[12];
    W[2] = (const float*)d_in[13]; G[2] = (const float*)d_in[15]; Be[2] = (const float*)d_in[16];
    W[3] = (const float*)d_in[17]; G[3] = (const float*)d_in[19]; Be[3] = (const float*)d_in[20];
    W[4] = (const float*)d_in[21]; G[4] = (const float*)d_in[23]; Be[4] = (const float*)d_in[24];
    G[5] = (const float*)d_in[25]; Be[5] = (const float*)d_in[26];
  }

  // Workspace layout (all offsets 256B aligned).
  auto alignup = [](size_t x) { return (x + 255) & ~(size_t)255; };
  char* base = (char*)d_ws;
  size_t o = 0;
  h16* Wh = (h16*)(base + o);                 // f16 weights (5 matrices)
  o = alignup(o + (size_t)(kC * kCin0 + 4 * kC * kC) * sizeof(h16));
  float* ss = (float*)(base + o);             // 6 stages of {scale,shift} x 256
  o = alignup(o + 6 * 2 * kC * sizeof(float));
  h16* X2 = (h16*)(base + o);                 // [16384, 352] f16
  o = alignup(o + (size_t)kM1 * kCin0 * sizeof(h16));
  h16* Xa = (h16*)(base + o);                 // [65536, 256] f16 ping
  o = alignup(o + (size_t)kM0 * kC * sizeof(h16));
  h16* Xb = (h16*)(base + o);                 // [65536, 256] f16 pong
  o = alignup(o + (size_t)kM0 * kC * sizeof(h16));
  float* Y = (float*)(base + o);              // [65536, 256] f32 GEMM output
  (void)ws_size; (void)out_size; (void)n_in;

  h16* Wh0 = Wh;
  h16* Wh1 = Wh0 + (size_t)kC * kCin0;
  h16* Wh2 = Wh1 + (size_t)kC * kC;
  h16* Wh3 = Wh2 + (size_t)kC * kC;
  h16* Wh4 = Wh3 + (size_t)kC * kC;

  // 0) weights -> f16
  f32_to_f16_kernel<<<(kC * kCin0 + 255) / 256, 256, 0, stream>>>(W[0], Wh0, kC * kCin0);
  f32_to_f16_kernel<<<(kC * kC + 255) / 256, 256, 0, stream>>>(W[1], Wh1, kC * kC);
  f32_to_f16_kernel<<<(kC * kC + 255) / 256, 256, 0, stream>>>(W[2], Wh2, kC * kC);
  f32_to_f16_kernel<<<(kC * kC + 255) / 256, 256, 0, stream>>>(W[3], Wh3, kC * kC);
  f32_to_f16_kernel<<<(kC * kC + 255) / 256, 256, 0, stream>>>(W[4], Wh4, kC * kC);

  // 1) fp2 interpolation -> X2 [16384, 352]
  interp2_kernel<<<(kM1 + 255) / 256, 256, 0, stream>>>(
      l1_xyz, l2_xyz, l1_points, l2_points, X2);

  const int ew1 = (kM1 * kC + 255) / 256;       // elementwise grids
  const int ew0 = (kM0 * kC + 255) / 256;
  const size_t lds352 = (size_t)kC * kCin0 * sizeof(h16);  // 176KB
  const size_t lds256 = (size_t)kC * kC * sizeof(h16);     // 128KB

  // 2) fp2 conv0: [16384,352]x[352,256]; BN+ReLU -> Xa
  gemm_wmma_f16_kernel<<<kM1 / 128, 256, lds352, stream>>>(X2, Wh0, Y, kM1, kCin0);
  bn_stats_kernel<<<kC, 256, 0, stream>>>(Y, G[0], Be[0], ss + 0 * 2 * kC, kM1);
  bn_relu_f16_kernel<<<ew1, 256, 0, stream>>>(Y, ss + 0 * 2 * kC, Xa, kM1);

  // 3) fp2 conv1: 256->256; BN+ReLU -> Xb (= l1p features, row-major f16)
  gemm_wmma_f16_kernel<<<kM1 / 128, 256, lds256, stream>>>(Xa, Wh1, Y, kM1, kC);
  bn_stats_kernel<<<kC, 256, 0, stream>>>(Y, G[1], Be[1], ss + 1 * 2 * kC, kM1);
  bn_relu_f16_kernel<<<ew1, 256, 0, stream>>>(Y, ss + 1 * 2 * kC, Xb, kM1);

  // 4) fp1 interpolation: Xb (l1p) -> Xa = X0 [65536, 256]
  interp1_kernel<<<(kM0 + 255) / 256, 256, 0, stream>>>(l0_xyz, l1_xyz, Xb, Xa);

  // 5) fp1 conv0
  gemm_wmma_f16_kernel<<<kM0 / 128, 256, lds256, stream>>>(Xa, Wh2, Y, kM0, kC);
  bn_stats_kernel<<<kC, 256, 0, stream>>>(Y, G[2], Be[2], ss + 2 * 2 * kC, kM0);
  bn_relu_f16_kernel<<<ew0, 256, 0, stream>>>(Y, ss + 2 * 2 * kC, Xb, kM0);

  // 6) fp1 conv1
  gemm_wmma_f16_kernel<<<kM0 / 128, 256, lds256, stream>>>(Xb, Wh3, Y, kM0, kC);
  bn_stats_kernel<<<kC, 256, 0, stream>>>(Y, G[3], Be[3], ss + 3 * 2 * kC, kM0);
  bn_relu_f16_kernel<<<ew0, 256, 0, stream>>>(Y, ss + 3 * 2 * kC, Xa, kM0);

  // 7) fp1 conv2 -> BN+ReLU (f32, in-place) -> final BN -> d_out
  gemm_wmma_f16_kernel<<<kM0 / 128, 256, lds256, stream>>>(Xa, Wh4, Y, kM0, kC);
  bn_stats_kernel<<<kC, 256, 0, stream>>>(Y, G[4], Be[4], ss + 4 * 2 * kC, kM0);
  bn_relu_f32_kernel<<<ew0, 256, 0, stream>>>(Y, ss + 4 * 2 * kC, Y, kM0);
  bn_stats_kernel<<<kC, 256, 0, stream>>>(Y, G[5], Be[5], ss + 5 * 2 * kC, kM0);
  bn_final_kernel<<<ew0, 256, 0, stream>>>(Y, ss + 5 * 2 * kC, (float*)d_out, kM0);

  // 8) second output: l0_xyz transposed [B, N0, 3]
  float* out_xyz = (float*)d_out + (size_t)kM0 * kC;
  xyz_transpose_kernel<<<(kM0 * 3 + 255) / 256, 256, 0, stream>>>(l0_xyz, out_xyz);
}
